// GCN_11278584119813
// MI455X (gfx1250) — matrix operand order repaired and checked by reading, not compiled, not run on hardware
//
#include <hip/hip_runtime.h>

#define NN 100000   // nodes
#define NE 1600000  // edges
#define NF 256      // in features
#define NH 128      // hidden
#define NC 40       // classes

typedef __attribute__((ext_vector_type(16))) __bf16 v16bf;
typedef __attribute__((ext_vector_type(8)))  float  v8f;
typedef __attribute__((ext_vector_type(4)))  unsigned int v4u;
typedef __attribute__((ext_vector_type(8)))  int v8i;
typedef __attribute__((ext_vector_type(4)))  int v4i;

#if __has_builtin(__builtin_amdgcn_tensor_load_to_lds) && \
    __has_builtin(__builtin_amdgcn_s_wait_tensorcnt)
#define USE_TDM 1
#endif

union AFrag { v16bf v; unsigned short s[16]; unsigned int u[8]; };

__device__ __forceinline__ unsigned short f2bf(float f) {
  union { float f; unsigned u; } a; a.f = f;
  unsigned u = a.u;
  return (unsigned short)((u + 0x7FFFu + ((u >> 16) & 1u)) >> 16);  // RNE
}

// ---------------- workspace layout (bytes) ----------------
// W1 swizzled bf16 fragments : 8 kchunks * 8 ntiles * 32 lanes * 16 halfs
#define W1S_ELEMS (8*8*32*16)          // 32768 halfs = 64 KB
// W2 swizzled bf16 fragments (padded 128x48): 4 * 3 * 32 * 16
#define W2S_ELEMS (4*3*32*16)          // 6144 halfs = 12 KB
static constexpr size_t OFF_W1S  = 0;
static constexpr size_t OFF_W2S  = 65536;
static constexpr size_t OFF_H0   = 131072;                         // 100000*128*4 (reused as h2)
static constexpr size_t OFF_AGG1 = OFF_H0 + (size_t)NN*NH*4;       // 100000*128*4
static constexpr size_t OFF_HB   = OFF_AGG1 + (size_t)NN*NH*4;     // 100000*128*2 bf16
// total ~128.2 MB

// ---------------- init: zero accumulators, build swizzled bf16 weights ----------------
__global__ __launch_bounds__(256) void k_init(const float* __restrict__ W1,
                                              const float* __restrict__ W2,
                                              unsigned short* __restrict__ W1s,
                                              unsigned short* __restrict__ W2s,
                                              float* __restrict__ agg1,
                                              float* __restrict__ dout) {
  unsigned i = blockIdx.x * 256u + threadIdx.x;
  const unsigned Z1 = NN * NH;   // 12.8M
  const unsigned Z2 = NN * NC;   // 4M
  if (i < Z1) { agg1[i] = 0.f; return; }
  i -= Z1;
  if (i < Z2) { dout[i] = 0.f; return; }
  i -= Z2;
  if (i < W1S_ELEMS) {
    // B-fragment layout: b[e] = B[k0*32 + 8*(lane>>4) + e + (e&8)][n*16 + (lane&15)]
    unsigned e = i & 15u, lane = (i >> 4) & 31u, n = (i >> 9) & 7u, k0 = i >> 12;
    unsigned col = n * 16u + (lane & 15u);
    unsigned K   = k0 * 32u + 8u * (lane >> 4) + e + (e & 8u);
    W1s[i] = f2bf(W1[(size_t)K * NH + col]);
    return;
  }
  i -= W1S_ELEMS;
  if (i < W2S_ELEMS) {
    unsigned e = i & 15u, lane = (i >> 4) & 31u, rest = i >> 9;
    unsigned n = rest % 3u, k0 = rest / 3u;
    unsigned col = n * 16u + (lane & 15u);
    unsigned K   = k0 * 32u + 8u * (lane >> 4) + e + (e & 8u);
    W2s[i] = f2bf(col < NC ? W2[(size_t)K * NC + col] : 0.f);
  }
}

// ---------------- GEMM1: h0[100000x128] = x[100000x256](f32->bf16) @ W1 ----------------
// 256 threads = 8 waves; each wave: 16 rows x 128 cols; K-loop 8 x 32.
__global__ __launch_bounds__(256) void k_gemm1(const float* __restrict__ x,
                                               const unsigned short* __restrict__ W1s,
                                               float* __restrict__ h0) {
  __shared__ __align__(16) unsigned short lds[W1S_ELEMS];  // 64 KB of WGP's 320 KB

#ifdef USE_TDM
  // One wave issues a single Tensor-Data-Mover DMA: 8192 x 8B line, global -> LDS.
  if (threadIdx.x < 32) {
    unsigned long long ga = (unsigned long long)(uintptr_t)W1s;
    unsigned ldsOff = (unsigned)(uintptr_t)(&lds[0]);  // flat->LDS: low 32 bits
    v4u g0;
    g0.x = 1u;                                          // count=1 (valid user D#)
    g0.y = ldsOff;                                      // lds_addr
    g0.z = (unsigned)(ga & 0xFFFFFFFFu);                // global_addr[31:0]
    g0.w = (unsigned)((ga >> 32) & 0x01FFFFFFu) | (2u << 30);  // addr[56:32] | type=2
    v8i g1;
    g1[0] = (int)(3u << 16);                            // wg_mask=0, data_size=3 (8B)
    g1[1] = (int)(0x2000u << 16);                       // tensor_dim0[15:0]=8192
    g1[2] = (int)(1u << 16);                            // tensor_dim0[31:16]=0 | tensor_dim1=1
    g1[3] = (int)(0x2000u << 16);                       // tensor_dim1 hi=0 | tile_dim0=8192
    g1[4] = 1;                                          // tile_dim1=1, tile_dim2=0
    g1[5] = 8192;                                       // tensor_dim0_stride lo
    g1[6] = 0;                                          // stride hi | dim1_stride lo
    g1[7] = 0;
    v4i gz4 = {0, 0, 0, 0};                             // groups 2/3 unused (2D)
    v8i gz8 = {0, 0, 0, 0, 0, 0, 0, 0};
    __builtin_amdgcn_tensor_load_to_lds(g0, g1, gz4, gz4, gz8, 0);
    __builtin_amdgcn_s_wait_tensorcnt(0);
  }
#else
  {
    const uint4* g = (const uint4*)W1s;
    uint4*       s = (uint4*)lds;
    #pragma unroll
    for (int i = 0; i < 16; ++i) s[threadIdx.x + 256 * i] = g[threadIdx.x + 256 * i];
  }
#endif
  __syncthreads();

  const int wave = threadIdx.x >> 5, lane = threadIdx.x & 31;
  const int hl = lane >> 4, m = lane & 15;          // half-wave, row-in-tile
  const int rowBase = blockIdx.x * 128 + wave * 16;
  int arow = rowBase + m; if (arow >= NN) arow = NN - 1;   // clamp loads, guard stores
  const float* xr = x + (size_t)arow * NF;

  v8f acc[8] = {};
  #pragma unroll
  for (int k0 = 0; k0 < 8; ++k0) {
    // A 16-bit layout: lane half hl holds K = kb+8*hl+{0..7} and kb+16+8*hl+{0..7}
    AFrag a;
    const float* p = xr + k0 * 32 + 8 * hl;
    float4 f0 = *(const float4*)(p);
    float4 f1 = *(const float4*)(p + 4);
    float4 f2 = *(const float4*)(p + 16);
    float4 f3 = *(const float4*)(p + 20);
    a.s[0]=f2bf(f0.x); a.s[1]=f2bf(f0.y); a.s[2]=f2bf(f0.z); a.s[3]=f2bf(f0.w);
    a.s[4]=f2bf(f1.x); a.s[5]=f2bf(f1.y); a.s[6]=f2bf(f1.z); a.s[7]=f2bf(f1.w);
    a.s[8]=f2bf(f2.x); a.s[9]=f2bf(f2.y); a.s[10]=f2bf(f2.z); a.s[11]=f2bf(f2.w);
    a.s[12]=f2bf(f3.x); a.s[13]=f2bf(f3.y); a.s[14]=f2bf(f3.z); a.s[15]=f2bf(f3.w);
    #pragma unroll
    for (int n = 0; n < 8; ++n) {
      v16bf b = *(const v16bf*)&lds[((k0 * 8 + n) * 32 + lane) * 16];
      acc[n] = __builtin_amdgcn_wmma_f32_16x16x32_bf16(
          false, a.v, false, b, (short)0, acc[n], false, false);
    }
  }
  // C/D layout: lanes 0-15 N=lane,M=r; lanes 16-31 N=lane-16,M=8+r
  #pragma unroll
  for (int n = 0; n < 8; ++n) {
    #pragma unroll
    for (int r = 0; r < 8; ++r) {
      int row = rowBase + hl * 8 + r;
      if (row < NN) h0[(size_t)row * NH + n * 16 + m] = acc[n][r];
    }
  }
}

// ---------------- SpMM1: agg1[dst] += w * h0[src]   (one wave per edge) ----------------
// e is wave-uniform; readfirstlane forces src/dst/w into SGPRs (s_load), so the
// 12B of edge metadata is fetched once per wave instead of once per lane.
__global__ __launch_bounds__(256) void k_spmm1(const int* __restrict__ src,
                                               const int* __restrict__ dst,
                                               const float* __restrict__ ew,
                                               const float* __restrict__ h0,
                                               float* __restrict__ agg1) {
  unsigned e = (unsigned)__builtin_amdgcn_readfirstlane(
      (int)(blockIdx.x * 8u + (threadIdx.x >> 5)));   // NE*32 divides grid exactly
  unsigned c = threadIdx.x & 31u;
  int s = src[e], d = dst[e];
  float w = ew[e];
  float4 v = *(const float4*)&h0[(size_t)s * NH + c * 4];
  float* p = &agg1[(size_t)d * NH + c * 4];
  atomicAdd(p + 0, v.x * w);
  atomicAdd(p + 1, v.y * w);
  atomicAdd(p + 2, v.z * w);
  atomicAdd(p + 3, v.w * w);
}

// ---------------- bias + relu + pack to bf16 ----------------
__global__ __launch_bounds__(256) void k_biasrelu(const float* __restrict__ agg1,
                                                  const float* __restrict__ b1,
                                                  unsigned short* __restrict__ hb) {
  unsigned i = blockIdx.x * 256u + threadIdx.x;
  if (i >= (unsigned)(NN * NH)) return;
  float v = agg1[i] + b1[i & 127u];
  hb[i] = f2bf(fmaxf(v, 0.f));
}

// ---------------- GEMM2: h2[100000x40] = hb[100000x128] @ W2 (padded N=48) ----------------
__global__ __launch_bounds__(256) void k_gemm2(const unsigned short* __restrict__ hb,
                                               const unsigned short* __restrict__ W2s,
                                               float* __restrict__ h2) {
  __shared__ __align__(16) unsigned short lds[W2S_ELEMS];  // 12 KB
  {
    const uint4* g = (const uint4*)W2s;
    uint4*       s = (uint4*)lds;
    for (int i = threadIdx.x; i < W2S_ELEMS / 8; i += 256) s[i] = g[i];
  }
  __syncthreads();

  const int wave = threadIdx.x >> 5, lane = threadIdx.x & 31;
  const int hl = lane >> 4, m = lane & 15;
  const int rowBase = blockIdx.x * 128 + wave * 16;
  int arow = rowBase + m; if (arow >= NN) arow = NN - 1;
  const unsigned short* hr = hb + (size_t)arow * NH;

  v8f acc[3] = {};
  #pragma unroll
  for (int k0 = 0; k0 < 4; ++k0) {
    AFrag a;
    const unsigned short* p = hr + k0 * 32 + 8 * hl;
    *(uint4*)&a.s[0] = *(const uint4*)(p);        // K rel 0..7   (8 bf16 = 16B)
    *(uint4*)&a.s[8] = *(const uint4*)(p + 16);   // K rel 16..23
    #pragma unroll
    for (int n = 0; n < 3; ++n) {
      v16bf b = *(const v16bf*)&lds[((k0 * 3 + n) * 32 + lane) * 16];
      acc[n] = __builtin_amdgcn_wmma_f32_16x16x32_bf16(
          false, a.v, false, b, (short)0, acc[n], false, false);
    }
  }
  #pragma unroll
  for (int n = 0; n < 3; ++n) {
    #pragma unroll
    for (int r = 0; r < 8; ++r) {
      int row = rowBase + hl * 8 + r;
      int col = n * 16 + m;
      if (row < NN && col < NC) h2[(size_t)row * NC + col] = acc[n][r];
    }
  }
}

// ---------------- SpMM2: dout[dst] += w * h2[src]   (40 feats, 4 per thread) ----------------
__global__ __launch_bounds__(256) void k_spmm2(const int* __restrict__ src,
                                               const int* __restrict__ dst,
                                               const float* __restrict__ ew,
                                               const float* __restrict__ h2,
                                               float* __restrict__ dout) {
  unsigned gid = blockIdx.x * 256u + threadIdx.x;
  unsigned e = gid / 10u, c = gid - e * 10u;
  if (e >= NE) return;
  int s = src[e], d = dst[e];
  float w = ew[e];
  float4 v = *(const float4*)&h2[(size_t)s * NC + c * 4];
  float* p = &dout[(size_t)d * NC + c * 4];
  atomicAdd(p + 0, v.x * w);
  atomicAdd(p + 1, v.y * w);
  atomicAdd(p + 2, v.z * w);
  atomicAdd(p + 3, v.w * w);
}

// ---------------- bias + log_softmax, in place on d_out. One wave32 per row. ----------------
__global__ __launch_bounds__(256) void k_lsm(float* __restrict__ dout,
                                             const float* __restrict__ b2) {
  const int wave = threadIdx.x >> 5, lane = threadIdx.x & 31;
  int row = blockIdx.x * 8 + wave;
  if (row >= NN) return;
  float* r = dout + (size_t)row * NC;
  float v1 = r[lane] + b2[lane];
  float v2 = (lane < 8) ? (r[32 + lane] + b2[32 + lane]) : -3.4e38f;
  float mx = fmaxf(v1, v2);
  #pragma unroll
  for (int off = 16; off; off >>= 1) mx = fmaxf(mx, __shfl_xor(mx, off, 32));
  float sum = __expf(v1 - mx) + ((lane < 8) ? __expf(v2 - mx) : 0.f);
  #pragma unroll
  for (int off = 16; off; off >>= 1) sum += __shfl_xor(sum, off, 32);
  float l = __logf(sum);
  r[lane] = v1 - mx - l;
  if (lane < 8) r[32 + lane] = v2 - mx - l;
}

extern "C" void kernel_launch(void* const* d_in, const int* in_sizes, int n_in,
                              void* d_out, int out_size, void* d_ws, size_t ws_size,
                              hipStream_t stream) {
  const float* x  = (const float*)d_in[0];
  const int*   ei = (const int*)d_in[1];
  const float* ew = (const float*)d_in[2];
  const float* W1 = (const float*)d_in[3];
  const float* b1 = (const float*)d_in[4];
  const float* W2 = (const float*)d_in[5];
  const float* b2 = (const float*)d_in[6];
  float* dout = (float*)d_out;

  char* ws = (char*)d_ws;
  unsigned short* W1s = (unsigned short*)(ws + OFF_W1S);
  unsigned short* W2s = (unsigned short*)(ws + OFF_W2S);
  float*          h0  = (float*)(ws + OFF_H0);
  float*          h2  = (float*)(ws + OFF_H0);    // aliases h0 (dead by then)
  float*          agg1= (float*)(ws + OFF_AGG1);
  unsigned short* hb  = (unsigned short*)(ws + OFF_HB);

  const int* src = ei;        // edge_index[0]
  const int* dst = ei + NE;   // edge_index[1]

  const unsigned initTotal = (unsigned)(NN * NH) + (unsigned)(NN * NC) + W1S_ELEMS + W2S_ELEMS;
  k_init    <<<(initTotal + 255) / 256, 256, 0, stream>>>(W1, W2, W1s, W2s, agg1, dout);
  k_gemm1   <<<(NN + 127) / 128, 256, 0, stream>>>(x, W1s, h0);
  k_spmm1   <<<((unsigned)NE * 32u) / 256u, 256, 0, stream>>>(src, dst, ew, h0, agg1);
  k_biasrelu<<<((unsigned)(NN * NH) + 255u) / 256u, 256, 0, stream>>>(agg1, b1, hb);
  k_gemm2   <<<(NN + 127) / 128, 256, 0, stream>>>(hb, W2s, h2);
  k_spmm2   <<<((unsigned)NE * 10u + 255u) / 256u, 256, 0, stream>>>(src, dst, ew, h2, dout);
  k_lsm     <<<(NN + 7) / 8, 256, 0, stream>>>(dout, b2);
}
